// encoder_77137612636815
// MI455X (gfx1250) — matrix operand order repaired and checked by reading, not compiled, not run on hardware
//
#include <hip/hip_runtime.h>

typedef __attribute__((ext_vector_type(16))) __bf16 v16bf;
typedef __attribute__((ext_vector_type(8)))  float  v8f;

#define CB   16
#define CL   2048
#define CD   512
#define CF   2048
#define CMN  (CB * CL)      // 32768 rows
#define CTOPK 7
#define TM   128
#define TN   128
#define TK   32
#define LPAD 40             // bf16 stride (32 + 8 pad), keeps 16B alignment

__device__ __forceinline__ unsigned short f2bf(float f) {
    unsigned int u = __float_as_uint(f);
    unsigned int r = u + 0x7FFFu + ((u >> 16) & 1u);   // round-to-nearest-even
    return (unsigned short)(r >> 16);
}

union Frag { v16bf v; uint4 q[2]; };
union Acc  { v8f   v; float f[8]; };

// Async global -> LDS copy (CDNA5, ASYNCcnt-tracked). 16 bytes per lane.
__device__ __forceinline__ void async_b128(unsigned lds_off, unsigned long long base,
                                           unsigned goff) {
    asm volatile("global_load_async_to_lds_b128 %0, %1, %2 offset:0"
                 :: "v"(lds_off), "v"(goff), "s"(base) : "memory");
}
__device__ __forceinline__ void wait_async0() {
    asm volatile("s_wait_asynccnt 0x0" ::: "memory");
}

// ---------------------------------------------------------------- fp32 -> bf16
__global__ __launch_bounds__(256)
void cvt_f32_bf16(const float* __restrict__ s, unsigned short* __restrict__ d, int n4) {
    int i = blockIdx.x * 256 + threadIdx.x;
    if (i >= n4) return;
    float4 f = ((const float4*)s)[i];
    ushort4 o;
    o.x = f2bf(f.x); o.y = f2bf(f.y); o.z = f2bf(f.z); o.w = f2bf(f.w);
    ((ushort4*)d)[i] = o;
}

// ----------------------------------------------- fp32 W[K][N] -> bf16 Wt[N][K]
__global__ __launch_bounds__(256)
void cvt_transpose_bf16(const float* __restrict__ s, unsigned short* __restrict__ d,
                        int K, int N) {
    int gid = blockIdx.x * 256 + threadIdx.x;
    if (gid >= K * N) return;
    int n = gid % N, k = gid / N;
    d[(size_t)n * K + k] = f2bf(s[gid]);
}

// ---------------------------------------------------------------- WMMA GEMM
// C[M,N] = A[M,K](bf16) @ Bt[N,K](bf16, pre-transposed)  (+bias)(+resid)(relu)
template<bool BIAS, bool RELU, bool RES, bool OUTF, bool OUTB>
__global__ __launch_bounds__(256)
void wmma_gemm(const unsigned short* __restrict__ A, const unsigned short* __restrict__ Bt,
               const float* __restrict__ bias, const float* __restrict__ resid,
               float* __restrict__ Cf, unsigned short* __restrict__ Cb,
               int M, int N, int K)
{
    __shared__ unsigned short lA[2][TM * LPAD];
    __shared__ unsigned short lB[2][TN * LPAD];   // [col][k], k contiguous
    const int tid  = threadIdx.x;
    const int lane = tid & 31;
    const int wave = tid >> 5;
    const int wm   = wave & 1;        // 64-row slab
    const int wn   = wave >> 1;       // 32-col slab
    const int m0   = blockIdx.y * TM;
    const int n0   = blockIdx.x * TN;
    const int lrow = lane & 15;
    const int kbA  = (lane < 16) ? 0 : 16;    // A halves: K 0..15 / 16..31 contiguous
    const int kbB  = (lane < 16) ? 0 : 16;    // B halves: K 0..15 / 16..31 contiguous

    const unsigned ldsA = (unsigned)(unsigned long long)(const void*)&lA[0][0];
    const unsigned ldsB = (unsigned)(unsigned long long)(const void*)&lB[0][0];
    const unsigned long long baseA = (unsigned long long)(const void*)A;
    const unsigned long long baseB = (unsigned long long)(const void*)Bt;

    auto issue = [&](int k0, int buf) {
        #pragma unroll
        for (int c = 0; c < 2; ++c) {
            int lin = tid + 256 * c;
            int r   = lin >> 2;
            int kc  = (lin & 3) << 3;
            unsigned lo = (unsigned)((buf * TM * LPAD + r * LPAD + kc) * 2);
            async_b128(ldsA + lo,
                       baseA, (unsigned)(((size_t)(m0 + r) * K + k0 + kc) * 2));
            async_b128(ldsB + lo,
                       baseB, (unsigned)(((size_t)(n0 + r) * K + k0 + kc) * 2));
        }
    };

    v8f vzero = {0.f,0.f,0.f,0.f,0.f,0.f,0.f,0.f};
    Acc acc[4][2];
    #pragma unroll
    for (int i = 0; i < 4; ++i)
        #pragma unroll
        for (int j = 0; j < 2; ++j) acc[i][j].v = vzero;

    issue(0, 0);
    wait_async0();
    __syncthreads();

    const int nk = K / TK;
    for (int kt = 0; kt < nk; ++kt) {
        const int cur = kt & 1;
        if (kt + 1 < nk) issue((kt + 1) * TK, cur ^ 1);
        #pragma unroll
        for (int j = 0; j < 2; ++j) {
            Frag bf;
            int col = wn * 32 + j * 16 + lrow;
            bf.q[0] = *(const uint4*)(&lB[cur][col * LPAD + kbB]);
            bf.q[1] = *(const uint4*)(&lB[cur][col * LPAD + kbB + 8]);
            #pragma unroll
            for (int i = 0; i < 4; ++i) {
                Frag af;
                int row = wm * 64 + i * 16 + lrow;
                af.q[0] = *(const uint4*)(&lA[cur][row * LPAD + kbA]);
                af.q[1] = *(const uint4*)(&lA[cur][row * LPAD + kbA + 8]);
                acc[i][j].v = __builtin_amdgcn_wmma_f32_16x16x32_bf16(
                    false, af.v, false, bf.v, (short)0, acc[i][j].v, false, false);
            }
        }
        wait_async0();
        __syncthreads();
    }

    const int cb = lane & 15;
    const int rh = (lane < 16) ? 0 : 8;
    #pragma unroll
    for (int i = 0; i < 4; ++i)
        #pragma unroll
        for (int j = 0; j < 2; ++j) {
            int gc = n0 + wn * 32 + j * 16 + cb;
            float bsv = BIAS ? bias[gc] : 0.f;
            #pragma unroll
            for (int r = 0; r < 8; ++r) {
                int gr = m0 + wm * 64 + i * 16 + rh + r;
                size_t o = (size_t)gr * N + gc;
                float val = acc[i][j].f[r] + bsv;
                if (RES)  val += resid[o];
                if (RELU) val = val > 0.f ? val : 0.f;
                if (OUTF) Cf[o] = val;
                if (OUTB) Cb[o] = f2bf(val);
            }
        }
}

// ------------------------------------------------- autocorrelation via Q K^T
// mean_corr[b,tau] += (1/D) * sum over circular diagonal (qrow - krow == tau)
__global__ __launch_bounds__(256)
void corr_wmma(const unsigned short* __restrict__ Q, const unsigned short* __restrict__ Km,
               float* __restrict__ mc)
{
    __shared__ unsigned short lA[2][TM * LPAD];
    __shared__ unsigned short lB[2][TN * LPAD];
    __shared__ float sbin[255];
    const int tid  = threadIdx.x;
    const int lane = tid & 31;
    const int wave = tid >> 5;
    const int wm   = wave & 1;
    const int wn   = wave >> 1;
    const int b    = blockIdx.z;
    const int m0   = blockIdx.y * TM;    // q time rows
    const int n0   = blockIdx.x * TN;    // k time rows
    const int lrow = lane & 15;
    const int kbA  = (lane < 16) ? 0 : 16;
    const int kbB  = (lane < 16) ? 0 : 16;

    const unsigned ldsA = (unsigned)(unsigned long long)(const void*)&lA[0][0];
    const unsigned ldsB = (unsigned)(unsigned long long)(const void*)&lB[0][0];
    const unsigned long long baseQ =
        (unsigned long long)(const void*)(Q + (size_t)b * CL * CD);
    const unsigned long long baseK =
        (unsigned long long)(const void*)(Km + (size_t)b * CL * CD);

    auto issue = [&](int k0, int buf) {
        #pragma unroll
        for (int c = 0; c < 2; ++c) {
            int lin = tid + 256 * c;
            int r   = lin >> 2;
            int kc  = (lin & 3) << 3;
            unsigned lo = (unsigned)((buf * TM * LPAD + r * LPAD + kc) * 2);
            async_b128(ldsA + lo, baseQ, (unsigned)(((size_t)(m0 + r) * CD + k0 + kc) * 2));
            async_b128(ldsB + lo, baseK, (unsigned)(((size_t)(n0 + r) * CD + k0 + kc) * 2));
        }
    };

    for (int i = tid; i < 255; i += 256) sbin[i] = 0.f;

    v8f vzero = {0.f,0.f,0.f,0.f,0.f,0.f,0.f,0.f};
    Acc acc[4][2];
    #pragma unroll
    for (int i = 0; i < 4; ++i)
        #pragma unroll
        for (int j = 0; j < 2; ++j) acc[i][j].v = vzero;

    issue(0, 0);
    wait_async0();
    __syncthreads();

    const int nk = CD / TK;
    for (int kt = 0; kt < nk; ++kt) {
        const int cur = kt & 1;
        if (kt + 1 < nk) issue((kt + 1) * TK, cur ^ 1);
        #pragma unroll
        for (int j = 0; j < 2; ++j) {
            Frag bf;
            int col = wn * 32 + j * 16 + lrow;
            bf.q[0] = *(const uint4*)(&lB[cur][col * LPAD + kbB]);
            bf.q[1] = *(const uint4*)(&lB[cur][col * LPAD + kbB + 8]);
            #pragma unroll
            for (int i = 0; i < 4; ++i) {
                Frag af;
                int row = wm * 64 + i * 16 + lrow;
                af.q[0] = *(const uint4*)(&lA[cur][row * LPAD + kbA]);
                af.q[1] = *(const uint4*)(&lA[cur][row * LPAD + kbA + 8]);
                acc[i][j].v = __builtin_amdgcn_wmma_f32_16x16x32_bf16(
                    false, af.v, false, bf.v, (short)0, acc[i][j].v, false, false);
            }
        }
        wait_async0();
        __syncthreads();
    }

    const int cb = lane & 15;
    const int rh = (lane < 16) ? 0 : 8;
    #pragma unroll
    for (int i = 0; i < 4; ++i)
        #pragma unroll
        for (int j = 0; j < 2; ++j) {
            int lc = wn * 32 + j * 16 + cb;
            #pragma unroll
            for (int r = 0; r < 8; ++r) {
                int lr = wm * 64 + i * 16 + rh + r;
                atomicAdd(&sbin[lr - lc + 127], acc[i][j].f[r]);
            }
        }
    __syncthreads();
    for (int i = tid; i < 255; i += 256) {
        int tau = (m0 - n0 + i - 127) & (CL - 1);
        atomicAdd(&mc[b * CL + tau], sbin[i] * (1.f / (float)CD));
    }
}

// ------------------------------------------------- top-7 over batch-mean corr
__global__ __launch_bounds__(256)
void topk_kernel(const float* __restrict__ mc, int* __restrict__ idx_out)
{
    __shared__ float sv[CL];
    __shared__ float smax[256];
    __shared__ int   sidx[256];
    int tid = threadIdx.x;
    for (int l = tid; l < CL; l += 256) {
        float a = 0.f;
        for (int b = 0; b < CB; ++b) a += mc[b * CL + l];
        sv[l] = a;
    }
    __syncthreads();
    for (int t = 0; t < CTOPK; ++t) {
        float best = -3.4e38f; int bi = 0;
        for (int l = tid; l < CL; l += 256)
            if (sv[l] > best) { best = sv[l]; bi = l; }
        smax[tid] = best; sidx[tid] = bi;
        __syncthreads();
        for (int s = 128; s > 0; s >>= 1) {
            if (tid < s && smax[tid + s] > smax[tid]) {
                smax[tid] = smax[tid + s]; sidx[tid] = sidx[tid + s];
            }
            __syncthreads();
        }
        if (tid == 0) { idx_out[t] = sidx[0]; sv[sidx[0]] = -3.4e38f; }
        __syncthreads();
    }
}

// ------------------------------------------------- per-batch softmax over lags
__global__ __launch_bounds__(32)
void softmax_kernel(const float* __restrict__ mc, const int* __restrict__ idx,
                    float* __restrict__ w)
{
    int b = blockIdx.x;
    __shared__ float vals[CTOPK];
    if (threadIdx.x < CTOPK) vals[threadIdx.x] = mc[b * CL + idx[threadIdx.x]];
    __syncthreads();
    if (threadIdx.x == 0) {
        float m = vals[0];
        for (int i = 1; i < CTOPK; ++i) m = fmaxf(m, vals[i]);
        float e[CTOPK], s = 0.f;
        for (int i = 0; i < CTOPK; ++i) { e[i] = __expf(vals[i] - m); s += e[i]; }
        for (int i = 0; i < CTOPK; ++i) w[b * 8 + i] = e[i] / s;
    }
}

// ------------------------------------------------- weighted circular gather
__global__ __launch_bounds__(256)
void agg_kernel(const float* __restrict__ v, const float* __restrict__ w,
                const int* __restrict__ idx, unsigned short* __restrict__ outb)
{
    const int chunks = CD / 4;
    int gid = blockIdx.x * 256 + threadIdx.x;
    int total = CB * CL * chunks;
    if (gid >= total) return;
    int dc = gid % chunks;
    int bl = gid / chunks;
    int l  = bl & (CL - 1);
    int b  = bl >> 11;
    float a0 = 0.f, a1 = 0.f, a2 = 0.f, a3 = 0.f;
    #pragma unroll
    for (int i = 0; i < CTOPK; ++i) {
        int src = (l + idx[i]) & (CL - 1);
        float4 vv = *(const float4*)(v + ((size_t)(b * CL + src) * CD) + dc * 4);
        float wi = w[b * 8 + i];
        a0 += wi * vv.x; a1 += wi * vv.y; a2 += wi * vv.z; a3 += wi * vv.w;
    }
    ushort4 o; o.x = f2bf(a0); o.y = f2bf(a1); o.z = f2bf(a2); o.w = f2bf(a3);
    ((ushort4*)outb)[gid] = o;
}

// ------------------------------------------------- series decomp (k=25, SAME, counted)
template<bool OUTB>
__global__ __launch_bounds__(256)
void decomp_kernel(const float* __restrict__ X, unsigned short* __restrict__ ob,
                   float* __restrict__ of, int Nc)
{
    size_t gid = (size_t)blockIdx.x * 256 + threadIdx.x;
    size_t total = (size_t)CMN * Nc;
    if (gid >= total) return;
    int n = (int)(gid % Nc);
    size_t bl = gid / Nc;
    int l = (int)(bl & (CL - 1));
    size_t brow0 = (bl - (size_t)l) * (size_t)Nc;
    int lo = l - 12; if (lo < 0) lo = 0;
    int hi = l + 12; if (hi > CL - 1) hi = CL - 1;
    float s = 0.f;
    const float* base = X + brow0 + n;
    for (int t = lo; t <= hi; ++t) s += base[(size_t)t * Nc];
    float val = X[gid] - s / (float)(hi - lo + 1);
    if (OUTB) ob[gid] = f2bf(val);
    else      of[gid] = val;
}

// ---------------------------------------------------------------- host launch
extern "C" void kernel_launch(void* const* d_in, const int* in_sizes, int n_in,
                              void* d_out, int out_size, void* d_ws, size_t ws_size,
                              hipStream_t stream)
{
    const float* x   = (const float*)d_in[0];
    const float* Win = (const float*)d_in[1];
    const float* bin = (const float*)d_in[2];
    const float* Wq  = (const float*)d_in[3];
    const float* bq  = (const float*)d_in[4];
    const float* Wk  = (const float*)d_in[5];
    const float* bk  = (const float*)d_in[6];
    const float* Wv  = (const float*)d_in[7];
    const float* bv  = (const float*)d_in[8];
    const float* Wo  = (const float*)d_in[9];
    const float* bo  = (const float*)d_in[10];
    const float* W1  = (const float*)d_in[11];
    const float* W2  = (const float*)d_in[12];
    float* out = (float*)d_out;
    char*  ws  = (char*)d_ws;

    const size_t SZ_BF = (size_t)CMN * CD * 2;   // 32 MB bf16 activation
    const size_t SZ_F  = (size_t)CMN * CD * 4;   // 64 MB fp32 activation

    unsigned short* xbf = (unsigned short*)(ws + 0);
    unsigned short* hbf = (unsigned short*)(ws + 1 * SZ_BF);
    unsigned short* qbf = (unsigned short*)(ws + 2 * SZ_BF);
    unsigned short* kbf = (unsigned short*)(ws + 3 * SZ_BF);
    float*          vfp = (float*)         (ws + 4 * SZ_BF);
    unsigned short* agb = (unsigned short*)(ws + 4 * SZ_BF + SZ_F);
    float*          xr  = (float*)         (ws + 5 * SZ_BF + SZ_F);
    float*          y2  = (float*)         (ws + 0);            // alias: early bufs dead
    unsigned short* sbf = (unsigned short*)(ws + 5 * SZ_BF + 2 * SZ_F);
    unsigned short* y1  = (unsigned short*)(ws + 6 * SZ_BF + 2 * SZ_F);
    size_t o = 6 * SZ_BF + 2 * SZ_F + (size_t)CMN * CF * 2;
    unsigned short* wib = (unsigned short*)(ws + o); o += (size_t)CD * CD * 2;
    unsigned short* wqb = (unsigned short*)(ws + o); o += (size_t)CD * CD * 2;
    unsigned short* wkb = (unsigned short*)(ws + o); o += (size_t)CD * CD * 2;
    unsigned short* wvb = (unsigned short*)(ws + o); o += (size_t)CD * CD * 2;
    unsigned short* wob = (unsigned short*)(ws + o); o += (size_t)CD * CD * 2;
    unsigned short* w1b = (unsigned short*)(ws + o); o += (size_t)CD * CF * 2;
    unsigned short* w2b = (unsigned short*)(ws + o); o += (size_t)CF * CF * 2;
    float*          mc  = (float*)         (ws + o); o += (size_t)CB * CL * 4;
    int*            idx = (int*)           (ws + o); o += 64;
    float*          wts = (float*)         (ws + o);

    const int T = 256;
    // activation conversion + weight transpose-conversions (Wt[N][K] bf16)
    cvt_f32_bf16<<<(CMN * CD / 4 + T - 1) / T, T, 0, stream>>>(x, xbf, CMN * CD / 4);
    cvt_transpose_bf16<<<(CD * CD + T - 1) / T, T, 0, stream>>>(Win, wib, CD, CD);
    cvt_transpose_bf16<<<(CD * CD + T - 1) / T, T, 0, stream>>>(Wq, wqb, CD, CD);
    cvt_transpose_bf16<<<(CD * CD + T - 1) / T, T, 0, stream>>>(Wk, wkb, CD, CD);
    cvt_transpose_bf16<<<(CD * CD + T - 1) / T, T, 0, stream>>>(Wv, wvb, CD, CD);
    cvt_transpose_bf16<<<(CD * CD + T - 1) / T, T, 0, stream>>>(Wo, wob, CD, CD);
    cvt_transpose_bf16<<<(CD * CF + T - 1) / T, T, 0, stream>>>(W1, w1b, CD, CF);
    cvt_transpose_bf16<<<(CF * CF + T - 1) / T, T, 0, stream>>>(W2, w2b, CF, CF);

    dim3 gD(CD / TN, CMN / TM);   // (4, 256)
    dim3 gF(CF / TN, CMN / TM);   // (16, 256)

    // h = x@Win + b_in ; q/k/v projections
    wmma_gemm<true, false, false, false, true><<<gD, T, 0, stream>>>(xbf, wib, bin, nullptr, nullptr, hbf, CMN, CD, CD);
    wmma_gemm<true, false, false, false, true><<<gD, T, 0, stream>>>(hbf, wqb, bq, nullptr, nullptr, qbf, CMN, CD, CD);
    wmma_gemm<true, false, false, false, true><<<gD, T, 0, stream>>>(hbf, wkb, bk, nullptr, nullptr, kbf, CMN, CD, CD);
    wmma_gemm<true, false, false, true, false><<<gD, T, 0, stream>>>(hbf, wvb, bv, nullptr, vfp, nullptr, CMN, CD, CD);

    // autocorrelation: diag-sums of Q K^T, top-7 lags, softmax, gather
    hipMemsetAsync(mc, 0, (size_t)CB * CL * 4, stream);
    corr_wmma<<<dim3(CL / TN, CL / TM, CB), T, 0, stream>>>(qbf, kbf, mc);
    topk_kernel<<<1, 256, 0, stream>>>(mc, idx);
    softmax_kernel<<<CB, 32, 0, stream>>>(mc, idx, wts);
    agg_kernel<<<(CB * CL * (CD / 4) + T - 1) / T, T, 0, stream>>>(vfp, wts, idx, agb);

    // xr = agg@Wo + bo + x ; s = decomp(xr) ; FFN ; final decomp
    wmma_gemm<true, false, true, true, false><<<gD, T, 0, stream>>>(agb, wob, bo, x, xr, nullptr, CMN, CD, CD);
    decomp_kernel<true><<<(int)(((size_t)CMN * CD + T - 1) / T), T, 0, stream>>>(xr, sbf, nullptr, CD);
    wmma_gemm<false, true, false, false, true><<<gF, T, 0, stream>>>(sbf, w1b, nullptr, nullptr, nullptr, y1, CMN, CF, CD);
    wmma_gemm<false, false, false, true, false><<<gF, T, 0, stream>>>(y1, w2b, nullptr, nullptr, y2, nullptr, CMN, CF, CF);
    decomp_kernel<false><<<(int)(((size_t)CMN * CF + T - 1) / T), T, 0, stream>>>(y2, nullptr, out, CF);
}
